// Attention_23733989277998
// MI455X (gfx1250) — matrix operand order repaired and checked by reading, not compiled, not run on hardware
//
#include <hip/hip_runtime.h>

// ---------------------------------------------------------------------------
// CDNA5 (gfx1250) bf16 WMMA attention block, software-pipelined.
//   kernel 1: qkv = x @ w_attn + b_attn                  (bf16 out, ws)
//   kernel 2: flash attention with adjacency mask        (bf16 out, ws)
//   kernel 3: out  = a @ w_proj + b_proj                 (f32 out)
//   kernel 4: out += q @ w_proj1 + b_proj1               (accumulate)
// ---------------------------------------------------------------------------

typedef __bf16 bf16_t;
typedef __attribute__((ext_vector_type(16))) __bf16        v16bf;
typedef __attribute__((ext_vector_type(4)))  __bf16        v4bf;
typedef __attribute__((ext_vector_type(8)))  float         v8f;
typedef __attribute__((ext_vector_type(4)))  float         v4f;
typedef __attribute__((ext_vector_type(4)))  unsigned int  v4u;

union FragU { v16bf v; v4u q[2]; };

// A-matrix 16x32 bf16 fragment: lanes 0-15 row M=lane hold
// K 0-15, lanes 16-31 hold K 16-31 (K-contiguous, matches B), packed 2 per dword.
// rowPtr must be 16-byte aligned -> two ds_load_b128.
__device__ __forceinline__ v16bf load_frag_a(const bf16_t* rowPtr, int hi) {
  FragU f;
  const char* p = (const char*)rowPtr;
  f.q[0] = *(const v4u*)(p + hi * 32);
  f.q[1] = *(const v4u*)(p + hi * 32 + 16);
  return f.v;
}

// B-matrix 32x16 bf16 fragment: N = lane&15 (column of Bt), lanes 0-15 hold
// K=0..15, lanes 16-31 hold K=16..31, K-contiguous -> two ds_load_b128.
__device__ __forceinline__ v16bf load_frag_b(const bf16_t* colPtr, int hi) {
  FragU f;
  const char* p = (const char*)colPtr + hi * 32;
  f.q[0] = *(const v4u*)(p);
  f.q[1] = *(const v4u*)(p + 16);
  return f.v;
}

__device__ __forceinline__ v8f wmma_bf16(v16bf a, v16bf b, v8f c) {
  // (neg_a, A, neg_b, B, c_mod, C, reuse_a, reuse_b)
  return __builtin_amdgcn_wmma_f32_16x16x32_bf16(false, a, false, b,
                                                 (short)0, c, false, false);
}

// 4-wide global loads (b128 for f32 input, b64 for bf16 input)
__device__ __forceinline__ v4f load4(const float* p) { return *(const v4f*)p; }
__device__ __forceinline__ v4f load4(const bf16_t* p) {
  v4bf h = *(const v4bf*)p;
  return __builtin_convertvector(h, v4f);
}

// ---------------------------------------------------------------------------
// GEMM: out[M,N] = A[M,K] @ W[K,N] (+bias) (+= out if acc).
// Block tile 64x128, 8 waves; wave w owns 16 columns x four 16-row sub-tiles.
// Double-buffered LDS; next k-tile is fetched into registers behind the
// WMMAs of the current k-tile (one barrier per k-step).
// ---------------------------------------------------------------------------
template <typename TIN, typename TOUT>
__global__ __launch_bounds__(256) void gemm_wmma(
    const TIN* __restrict__ A, int lda,
    const float* __restrict__ W, int ldw,
    const float* __restrict__ bias,
    TOUT* __restrict__ out, int ldo,
    int K, int acc)
{
  constexpr int AP = 40, BP = 40;                 // 80-byte rows (16B aligned)
  __shared__ alignas(16) bf16_t Asub[2][64 * AP];
  __shared__ alignas(16) bf16_t Bt[2][128 * BP];  // W tile transposed

  const int t    = threadIdx.x;
  const int lane = t & 31, wave = t >> 5;
  const int m0   = blockIdx.x * 64;
  const int n0   = blockIdx.y * 128;
  const int hi   = lane >> 4;

  const int wn4 = (t & 31) * 4;                   // W micro-tile: 4 cols
  const int wk4 = (t >> 5) * 4;                   //               4 rows

  v4f aR[2], wR[4];

  auto load_regs = [&](int k0) {
#pragma unroll
    for (int i = 0; i < 2; ++i) {                 // A: 1x4 chunks
      int e = t + i * 256;
      int r = e >> 3, kk4 = (e & 7) * 4;
      aR[i] = load4(&A[(size_t)(m0 + r) * lda + k0 + kk4]);
    }
#pragma unroll
    for (int j = 0; j < 4; ++j)                   // W: 4x4 micro-tile (b128)
      wR[j] = load4(&W[(size_t)(k0 + wk4 + j) * ldw + n0 + wn4]);
  };

  auto store_lds = [&](int buf) {
#pragma unroll
    for (int i = 0; i < 2; ++i) {
      int e = t + i * 256;
      int r = e >> 3, kk4 = (e & 7) * 4;
      *(v4bf*)&Asub[buf][r * AP + kk4] = __builtin_convertvector(aR[i], v4bf);
    }
#pragma unroll
    for (int c2 = 0; c2 < 4; ++c2) {              // register transpose
      v4f col = {wR[0][c2], wR[1][c2], wR[2][c2], wR[3][c2]};
      *(v4bf*)&Bt[buf][(wn4 + c2) * BP + wk4] = __builtin_convertvector(col, v4bf);
    }
  };

  v8f c[4];
#pragma unroll
  for (int i = 0; i < 4; ++i)
    c[i] = v8f{0.f, 0.f, 0.f, 0.f, 0.f, 0.f, 0.f, 0.f};

  load_regs(0);
  store_lds(0);
  __syncthreads();

  int cur = 0;
  for (int k0 = 0; k0 < K; k0 += 32) {
    const bool more = (k0 + 32) < K;
    if (more) load_regs(k0 + 32);                 // prefetch next tile

    v16bf bfrag = load_frag_b(&Bt[cur][(wave * 16 + (lane & 15)) * BP], hi);
#pragma unroll
    for (int ms = 0; ms < 4; ++ms) {
      v16bf afrag = load_frag_a(&Asub[cur][(ms * 16 + (lane & 15)) * AP], hi);
      c[ms] = wmma_bf16(afrag, bfrag, c[ms]);
    }

    if (more) store_lds(cur ^ 1);                 // commit to other buffer
    __syncthreads();
    cur ^= 1;
  }

  // C/D layout: lanes 0-15 N=lane, VGPR r -> M=r; lanes 16-31 -> M=8+r.
  const int col = n0 + wave * 16 + (lane & 15);
  const float bv = bias ? bias[col] : 0.0f;
#pragma unroll
  for (int ms = 0; ms < 4; ++ms) {
#pragma unroll
    for (int r = 0; r < 8; ++r) {
      int m = m0 + ms * 16 + hi * 8 + r;
      size_t idx = (size_t)m * ldo + col;
      float v = c[ms][r] + bv;
      if (acc) v += (float)out[idx];
      out[idx] = (TOUT)v;
    }
  }
}

// ---------------------------------------------------------------------------
// Flash attention with adjacency mask (no scale, per reference).
// Block = 64 q-rows for one (b,h); 8 waves: m_sub = w&3 (16 q-rows),
// ngrp = w>>2 (32-col half). O accumulators live in registers and are fed
// back as the WMMA C operand. K/V blocks are double-buffered in LDS and
// prefetched behind the S/softmax/PV phases.
// ---------------------------------------------------------------------------
__global__ __launch_bounds__(256) void flash_attn(
    const bf16_t* __restrict__ qkv,   // [B*T, 3072] bf16 (q|k|v)
    const float*  __restrict__ adj,   // [T, T] f32
    bf16_t*       __restrict__ aout)  // [B*T, 1024] bf16 merged heads
{
  constexpr int T = 2048, NX = 1024, LD = 3 * NX;
  constexpr int TP = 72;              // 144-byte rows (16B aligned)
  constexpr int SP = 68;              // 272-byte rows (16B aligned)
  constexpr float NEG = -1.0e9f;

  __shared__ alignas(16) bf16_t Qs[64 * TP];       // Q tile [qrow][d]
  __shared__ alignas(16) bf16_t Ks[2][64 * TP];    // K tile [key][d]
  __shared__ alignas(16) bf16_t Vt[2][64 * TP];    // V tile transposed [d][key]
  __shared__ alignas(16) float  Sf[64 * SP];       // scores f32
  __shared__ float mrow[64], lrow[64], alf[64];
  bf16_t* Pb = (bf16_t*)Sf;           // P (bf16, stride TP) aliases S

  const int t    = threadIdx.x;
  const int lane = t & 31, wave = t >> 5;
  const int hi   = lane >> 4;
  const int b    = blockIdx.y >> 4;   // H = 16
  const int h    = blockIdx.y & 15;
  const int q0   = blockIdx.x * 64;

  const bf16_t* Qg = qkv + (size_t)b * T * LD + h * 64;
  const bf16_t* Kg = Qg + NX;
  const bf16_t* Vg = Qg + 2 * NX;

  // staging micro-tile: 16 lanes along d (4 each), 16 rows groups of 4
  const int d4  = (t & 15) * 4;
  const int r4  = (t >> 4) * 4;

#pragma unroll
  for (int j = 0; j < 4; ++j) {       // Q tile: b64 loads, b64 LDS stores
    v4bf qv = *(const v4bf*)&Qg[(size_t)(q0 + r4 + j) * LD + d4];
    *(v4bf*)&Qs[(r4 + j) * TP + d4] = qv;
  }
  if (t < 64) { mrow[t] = -3.0e38f; lrow[t] = 0.0f; }

  v4bf kR[4], vR[4];
  auto load_kv = [&](int k0) {
#pragma unroll
    for (int j = 0; j < 4; ++j) {
      kR[j] = *(const v4bf*)&Kg[(size_t)(k0 + r4 + j) * LD + d4];
      vR[j] = *(const v4bf*)&Vg[(size_t)(k0 + r4 + j) * LD + d4];
    }
  };
  auto store_kv = [&](int buf) {
#pragma unroll
    for (int j = 0; j < 4; ++j)
      *(v4bf*)&Ks[buf][(r4 + j) * TP + d4] = kR[j];
#pragma unroll
    for (int c2 = 0; c2 < 4; ++c2) {  // register transpose for V
      v4bf col = {vR[0][c2], vR[1][c2], vR[2][c2], vR[3][c2]};
      *(v4bf*)&Vt[buf][(d4 + c2) * TP + r4] = col;
    }
  };

  const int m_sub = wave & 3;
  const int ngrp  = wave >> 2;
  v8f o0 = {0.f, 0.f, 0.f, 0.f, 0.f, 0.f, 0.f, 0.f};
  v8f o1 = {0.f, 0.f, 0.f, 0.f, 0.f, 0.f, 0.f, 0.f};

  load_kv(0);
  store_kv(0);

  int cur = 0;
  constexpr int NB = T / 64;
  for (int kb = 0; kb < NB; ++kb) {
    const int k0 = kb * 64;
    __syncthreads();                  // tiles[cur] + Qs ready; Pb free

    if (kb + 1 < NB) load_kv(k0 + 64);   // prefetch next K/V block

    // ---- S = Q K^T (each wave: 2 column tiles, 2 k-steps) ----
    {
      v8f s0 = {0.f, 0.f, 0.f, 0.f, 0.f, 0.f, 0.f, 0.f};
      v8f s1 = {0.f, 0.f, 0.f, 0.f, 0.f, 0.f, 0.f, 0.f};
      const int qrow = m_sub * 16 + (lane & 15);
      const int ncol = ngrp * 32 + (lane & 15);
#pragma unroll
      for (int kk0 = 0; kk0 < 64; kk0 += 32) {
        v16bf a  = load_frag_a(&Qs[qrow * TP + kk0], hi);
        v16bf b0 = load_frag_b(&Ks[cur][ncol * TP + kk0], hi);
        v16bf b1 = load_frag_b(&Ks[cur][(ncol + 16) * TP + kk0], hi);
        s0 = wmma_bf16(a, b0, s0);
        s1 = wmma_bf16(a, b1, s1);
      }
      const int mloc = m_sub * 16 + hi * 8;
      const int c0   = ngrp * 32 + (lane & 15);
#pragma unroll
      for (int r = 0; r < 8; ++r) {
        Sf[(mloc + r) * SP + c0]      = s0[r];
        Sf[(mloc + r) * SP + c0 + 16] = s1[r];
      }
    }
    __syncthreads();

    // ---- mask + streaming softmax (thread t: row t>>2, 16-col quarter) ----
    const int rr = t >> 2, qq = t & 3;
    float sv[16];
    float pmax = -3.0e38f;
    const float* adjrow = adj + (size_t)(q0 + rr) * T + k0 + qq * 16;
#pragma unroll
    for (int jj = 0; jj < 4; ++jj) {
      v4f av = *(const v4f*)(adjrow + jj * 4);
#pragma unroll
      for (int c2 = 0; c2 < 4; ++c2) {
        int j = jj * 4 + c2;
        float s = Sf[rr * SP + qq * 16 + j];
        s = s * av[c2] + NEG * (1.0f - av[c2]);
        sv[j] = s;
        pmax = fmaxf(pmax, s);
      }
    }
    pmax = fmaxf(pmax, __shfl_xor(pmax, 1));
    pmax = fmaxf(pmax, __shfl_xor(pmax, 2));
    const float mold = mrow[rr];
    const float mnew = fmaxf(mold, pmax);
    const float arow = __expf(mold - mnew);
    __syncthreads();                  // all Sf reads done before Pb overwrite

    float psum = 0.0f;
#pragma unroll
    for (int jj = 0; jj < 4; ++jj) {
      v4bf pk;
#pragma unroll
      for (int c2 = 0; c2 < 4; ++c2) {
        float p = __expf(sv[jj * 4 + c2] - mnew);
        psum += p;
        pk[c2] = (bf16_t)p;
      }
      *(v4bf*)&Pb[rr * TP + qq * 16 + jj * 4] = pk;
    }
    psum += __shfl_xor(psum, 1);
    psum += __shfl_xor(psum, 2);
    if (qq == 0) {
      mrow[rr] = mnew;
      lrow[rr] = lrow[rr] * arow + psum;
      alf[rr]  = arow;
    }
    __syncthreads();                  // Pb + alf ready

    // ---- O = O*alpha + P @ V (O registers are the WMMA C operand) ----
    {
      const int mloc = m_sub * 16 + hi * 8;
#pragma unroll
      for (int r = 0; r < 8; ++r) {
        float a_ = alf[mloc + r];
        o0[r] = o0[r] * a_;
        o1[r] = o1[r] * a_;
      }
      const int qrow = m_sub * 16 + (lane & 15);
      const int dcol = ngrp * 32 + (lane & 15);
#pragma unroll
      for (int kk0 = 0; kk0 < 64; kk0 += 32) {
        v16bf a  = load_frag_a(&Pb[qrow * TP + kk0], hi);
        v16bf b0 = load_frag_b(&Vt[cur][dcol * TP + kk0], hi);
        v16bf b1 = load_frag_b(&Vt[cur][(dcol + 16) * TP + kk0], hi);
        o0 = wmma_bf16(a, b0, o0);
        o1 = wmma_bf16(a, b1, o1);
      }
    }

    if (kb + 1 < NB) store_kv(cur ^ 1);   // commit prefetched tiles
    cur ^= 1;
  }

  // ---- normalize and store merged-head output [B*T, 1024] ----
  {
    const int c0 = h * 64 + ngrp * 32 + (lane & 15);
#pragma unroll
    for (int r = 0; r < 8; ++r) {
      int m = m_sub * 16 + hi * 8 + r;
      float inv = 1.0f / lrow[m];
      size_t row = (size_t)(b * T + q0 + m) * NX;
      aout[row + c0]      = (bf16_t)(o0[r] * inv);
      aout[row + c0 + 16] = (bf16_t)(o1[r] * inv);
    }
  }
}

// ---------------------------------------------------------------------------
extern "C" void kernel_launch(void* const* d_in, const int* in_sizes, int n_in,
                              void* d_out, int out_size, void* d_ws, size_t ws_size,
                              hipStream_t stream) {
  (void)in_sizes; (void)n_in; (void)out_size; (void)ws_size;

  const float* x       = (const float*)d_in[0];
  const float* adj     = (const float*)d_in[1];
  const float* w_attn  = (const float*)d_in[2];
  const float* b_attn  = (const float*)d_in[3];
  const float* w_proj  = (const float*)d_in[4];
  const float* b_proj  = (const float*)d_in[5];
  const float* w_proj1 = (const float*)d_in[6];
  const float* b_proj1 = (const float*)d_in[7];

  constexpr int Bb = 2, T = 2048, NX = 1024, H = 16;
  constexpr int M  = Bb * T;                         // 4096 rows

  bf16_t* qkv  = (bf16_t*)d_ws;                      // [M, 3072] bf16 (24 MB)
  bf16_t* aatt = qkv + (size_t)M * 3 * NX;           // [M, 1024] bf16 ( 8 MB)
  float*  out  = (float*)d_out;

  // 1) fused QKV projection
  gemm_wmma<float, bf16_t><<<dim3(M / 64, (3 * NX) / 128), 256, 0, stream>>>(
      x, NX, w_attn, 3 * NX, b_attn, qkv, 3 * NX, NX, 0);

  // 2) masked flash attention (grid: q-tiles x (B*H))
  flash_attn<<<dim3(T / 64, Bb * H), 256, 0, stream>>>(qkv, adj, aatt);

  // 3) out = attn @ w_proj + b_proj
  gemm_wmma<bf16_t, float><<<dim3(M / 64, NX / 128), 256, 0, stream>>>(
      aatt, NX, w_proj, NX, b_proj, out, NX, NX, 0);

  // 4) out += q @ w_proj1 + b_proj1   (q = first NX columns of qkv, lda=3NX)
  gemm_wmma<bf16_t, float><<<dim3(M / 64, NX / 128), 256, 0, stream>>>(
      qkv, 3 * NX, w_proj1, NX, b_proj1, out, NX, NX, 1);
}
